// CNNSelfAttention_7086696038651
// MI455X (gfx1250) — matrix operand order repaired and checked by reading, not compiled, not run on hardware
//
#include <hip/hip_runtime.h>
#include <hip/hip_bf16.h>
#include <math.h>

// ---------------------------------------------------------------------------
// CNN self-attention (SAGAN) for MI455X / gfx1250, wave32 + WMMA bf16.
//   B=4, C=256, H=W=64 -> N=4096, Cqk=32
// All matmuls on the WMMA pipe:
//   1) convert_w:  fp32 weights -> bf16
//   2) transpose_x: x (B,C,N) fp32 -> xT (B,N,C) bf16  (LDS tiled)
//   3) proj_wmma:  q/k/v projections, 8 resident x B-frags x 20 output tiles
//   4) attn:       flash-attention, 64-key steps, DPP8 max-reduce,
//                  WMMA row-sums, 22 WMMAs per step per wave
// ---------------------------------------------------------------------------

typedef __attribute__((ext_vector_type(16))) __bf16 v16bf;
typedef __attribute__((ext_vector_type(8)))  __bf16 v8bf;
typedef __attribute__((ext_vector_type(8)))  float  v8f;

#define C_DIM 256
#define N_PIX 4096
#define CQK   32
#define B_DIM 4
#define LDP   72   // padded LDS row stride (bf16 elems) for the 16x64 P tile

#define DPP8_SEL(a,b,c,d,e,f,g,hh) \
    ((a)|((b)<<3)|((c)<<6)|((d)<<9)|((e)<<12)|((f)<<15)|((g)<<18)|((hh)<<21))
#define DPP8_XOR1 DPP8_SEL(1,0,3,2,5,4,7,6)
#define DPP8_XOR2 DPP8_SEL(2,3,0,1,6,7,4,5)
#define DPP8_XOR4 DPP8_SEL(4,5,6,7,0,1,2,3)

template <int SEL>
__device__ __forceinline__ float dpp8_f(float v) {
    return __uint_as_float(__builtin_amdgcn_mov_dpp8(__float_as_uint(v), SEL));
}

// max-reduce across the 16 lanes of a half-wave (lanes 0-15 / 16-31 separately)
__device__ __forceinline__ float halfwave_max16(float v) {
    v = fmaxf(v, dpp8_f<DPP8_XOR1>(v));
    v = fmaxf(v, dpp8_f<DPP8_XOR2>(v));
    v = fmaxf(v, dpp8_f<DPP8_XOR4>(v));
    v = fmaxf(v, __shfl_xor(v, 8, 32));
    return v;
}

// ---------------------------------------------------------------------------
// weights fp32 -> bf16
// ---------------------------------------------------------------------------
__global__ __launch_bounds__(256) void convert_w_kernel(
    const float* __restrict__ Wq, const float* __restrict__ Wk,
    const float* __restrict__ Wv,
    __bf16* __restrict__ Wqb, __bf16* __restrict__ Wkb, __bf16* __restrict__ Wvb)
{
    int i = blockIdx.x * 256 + threadIdx.x;          // grid covers 65536
    if (i < CQK * C_DIM) {
        Wqb[i] = (__bf16)Wq[i];
        Wkb[i] = (__bf16)Wk[i];
    }
    Wvb[i] = (__bf16)Wv[i];
}

// ---------------------------------------------------------------------------
// x (B,C,N) fp32 -> xT (B,N,C) bf16, LDS-tiled 32x32 transpose
// ---------------------------------------------------------------------------
__global__ __launch_bounds__(256) void transpose_x_kernel(
    const float* __restrict__ x, __bf16* __restrict__ xT)
{
    __shared__ float tile[32][33];
    const int tx = threadIdx.x & 31;
    const int ty = threadIdx.x >> 5;                 // 0..7
    const int n0 = blockIdx.x * 32;
    const int c0 = blockIdx.y * 32;
    const int b  = blockIdx.z;

    const float* xb = x + (size_t)b * C_DIM * N_PIX;
#pragma unroll
    for (int j = 0; j < 32; j += 8)                  // coalesced over n
        tile[ty + j][tx] = xb[(size_t)(c0 + ty + j) * N_PIX + n0 + tx];
    __syncthreads();

    __bf16* xtb = xT + (size_t)b * N_PIX * C_DIM;
#pragma unroll
    for (int j = 0; j < 32; j += 8)                  // coalesced over c
        xtb[(size_t)(n0 + ty + j) * C_DIM + c0 + tx] = (__bf16)tile[tx][ty + j];
}

// ---------------------------------------------------------------------------
// WMMA projections. Block = 8 waves; wave handles 16 pixels (cols) and sweeps
// all 20 output tiles (2 q + 2 k + 16 v), K = 256 = 8 x WMMA(16x16x32).
//   D[d, n] = sum_cc W[d, cc] * xT[n, cc]
// ---------------------------------------------------------------------------
__global__ __launch_bounds__(256) void proj_wmma_kernel(
    const __bf16* __restrict__ xT,
    const __bf16* __restrict__ Wqb, const float* __restrict__ bq,
    const __bf16* __restrict__ Wkb, const float* __restrict__ bk,
    const __bf16* __restrict__ Wvb, const float* __restrict__ bv,
    __bf16* __restrict__ qT, __bf16* __restrict__ kT, __bf16* __restrict__ vB)
{
    const int lane = threadIdx.x & 31;
    const int wave = threadIdx.x >> 5;
    const int l16  = lane & 15;
    const int h    = lane >> 4;
    const int b    = blockIdx.y;
    const int n0   = blockIdx.x * 128 + wave * 16;   // pixel tile base

    const v8f vzero = {};

    // resident x B-fragments: col = pixel n0+l16, K (cc) contiguous per half
    const __bf16* xrow = xT + ((size_t)b * N_PIX + n0 + l16) * C_DIM;
    v16bf xf[8];
#pragma unroll
    for (int t = 0; t < 8; ++t)
        xf[t] = *(const v16bf*)(xrow + 32 * t + 16 * h);

    // ---- q and k: 2 row-tiles each
#pragma unroll
    for (int r0 = 0; r0 < CQK; r0 += 16) {
        v8f aq = vzero, ak = vzero;
#pragma unroll
        for (int t = 0; t < 8; ++t) {
            const __bf16* wq = Wqb + (size_t)(r0 + l16) * C_DIM + 32 * t;
            const __bf16* wk = Wkb + (size_t)(r0 + l16) * C_DIM + 32 * t;
            v8bf qlo = *(const v8bf*)(wq + 16 * h);
            v8bf qhi = *(const v8bf*)(wq + 8 + 16 * h);
            v8bf klo = *(const v8bf*)(wk + 16 * h);
            v8bf khi = *(const v8bf*)(wk + 8 + 16 * h);
            v16bf wqa, wka;
#pragma unroll
            for (int i = 0; i < 8; ++i) {
                wqa[i] = qlo[i]; wqa[8 + i] = qhi[i];
                wka[i] = klo[i]; wka[8 + i] = khi[i];
            }
            aq = __builtin_amdgcn_wmma_f32_16x16x32_bf16(false, wqa, false, xf[t],
                                                         (short)0, aq, false, false);
            ak = __builtin_amdgcn_wmma_f32_16x16x32_bf16(false, wka, false, xf[t],
                                                         (short)0, ak, false, false);
        }
        // C-layout: (vgpr r, lane) -> d = r0+r+8h, n = n0+l16
        __bf16* qd = qT + ((size_t)b * N_PIX + n0 + l16) * CQK;
        __bf16* kd = kT + ((size_t)b * N_PIX + n0 + l16) * CQK;
#pragma unroll
        for (int r = 0; r < 8; ++r) {
            int d = r0 + r + 8 * h;
            qd[d] = (__bf16)(aq[r] + bq[d]);
            kd[d] = (__bf16)(ak[r] + bk[d]);
        }
    }

    // ---- v: 16 row-tiles of 16 channels
    for (int r0 = 0; r0 < C_DIM; r0 += 16) {
        v8f av = vzero;
#pragma unroll
        for (int t = 0; t < 8; ++t) {
            const __bf16* wv = Wvb + (size_t)(r0 + l16) * C_DIM + 32 * t;
            v8bf vlo = *(const v8bf*)(wv + 16 * h);
            v8bf vhi = *(const v8bf*)(wv + 8 + 16 * h);
            v16bf wva;
#pragma unroll
            for (int i = 0; i < 8; ++i) { wva[i] = vlo[i]; wva[8 + i] = vhi[i]; }
            av = __builtin_amdgcn_wmma_f32_16x16x32_bf16(false, wva, false, xf[t],
                                                         (short)0, av, false, false);
        }
#pragma unroll
        for (int r = 0; r < 8; ++r) {
            int c = r0 + r + 8 * h;
            vB[((size_t)b * C_DIM + c) * N_PIX + n0 + l16] = (__bf16)(av[r] + bv[c]);
        }
    }
}

// ---------------------------------------------------------------------------
// Flash attention: one block = 16 queries; 2 waves split C=256 into 128 each.
// ---------------------------------------------------------------------------
__global__ __launch_bounds__(64) void attn_kernel(
    const __bf16* __restrict__ qT,
    const __bf16* __restrict__ kT,
    const __bf16* __restrict__ vB,
    const float*  __restrict__ x,
    const float*  __restrict__ gamma,
    float* __restrict__ out)
{
    const int lane = threadIdx.x & 31;
    const int wave = threadIdx.x >> 5;
    const int l16  = lane & 15;
    const int h    = lane >> 4;          // half-wave selector
    const int b    = blockIdx.y;
    const int m0   = blockIdx.x * 16;    // query tile base

    __shared__ __align__(16) __bf16 pbuf[2 * 16 * LDP];
    __bf16* pw = &pbuf[wave * 16 * LDP]; // private per-wave 16x64 P staging tile

    const v8f vzero = {};

    // Q A-fragment (16 queries x 32 dims)
    const __bf16* qrow = qT + ((size_t)b * N_PIX + m0 + l16) * CQK;
    v8bf qlo = *(const v8bf*)(qrow + 16 * h);
    v8bf qhi = *(const v8bf*)(qrow + 8 + 16 * h);
    v16bf qa;
#pragma unroll
    for (int i = 0; i < 8; ++i) { qa[i] = qlo[i]; qa[8 + i] = qhi[i]; }

    // ones B-fragment for WMMA row-sum of P
    v16bf onesb;
#pragma unroll
    for (int i = 0; i < 16; ++i) onesb[i] = (__bf16)1.0f;

    v8f acc[8];
#pragma unroll
    for (int f = 0; f < 8; ++f) acc[f] = vzero;
    float rmax[8], rsum[8];
#pragma unroll
    for (int r = 0; r < 8; ++r) { rmax[r] = -__builtin_inff(); rsum[r] = 0.0f; }

    const int cbase = wave * 128;
    const __bf16* kbase = kT + (size_t)b * N_PIX * CQK;
    const __bf16* vbase = vB + (size_t)b * C_DIM * N_PIX;
    const float sc = 0.0625f;            // 1/sqrt(C) = 1/16

    for (int kb = 0; kb < N_PIX; kb += 64) {
        if (kb + 64 < N_PIX) {           // global_prefetch_b8 of next tiles
            __builtin_prefetch(kbase + (size_t)(kb + 64 + l16) * CQK, 0, 0);
            __builtin_prefetch(vbase + (size_t)(cbase + l16) * N_PIX + kb + 64, 0, 0);
        }

        // 4 QK WMMAs: S[t] covers keys kb+16t .. kb+16t+15
        v8f S[4];
#pragma unroll
        for (int t = 0; t < 4; ++t) {
            const __bf16* kp = kbase + (size_t)(kb + 16 * t + l16) * CQK + 16 * h;
            v16bf kf = *(const v16bf*)kp;
            S[t] = __builtin_amdgcn_wmma_f32_16x16x32_bf16(false, qa, false, kf,
                                                           (short)0, vzero,
                                                           false, false);
        }

        // online softmax (C-layout: vgpr r -> row r+8h, 16 lanes/row)
        float corr[8];
#pragma unroll
        for (int r = 0; r < 8; ++r) {
            float s0 = S[0][r] * sc, s1 = S[1][r] * sc;
            float s2 = S[2][r] * sc, s3 = S[3][r] * sc;
            float mloc = fmaxf(fmaxf(s0, s1), fmaxf(s2, s3));
            float mrow = halfwave_max16(mloc);
            float mnew = fmaxf(rmax[r], mrow);
            corr[r] = __expf(rmax[r] - mnew);
            rmax[r] = mnew;
            float p0 = __expf(s0 - mnew);
            float p1 = __expf(s1 - mnew);
            float p2 = __expf(s2 - mnew);
            float p3 = __expf(s3 - mnew);
            __bf16* prow = pw + (r + 8 * h) * LDP;
            prow[l16]      = (__bf16)p0;
            prow[16 + l16] = (__bf16)p1;
            prow[32 + l16] = (__bf16)p2;
            prow[48 + l16] = (__bf16)p3;
        }
#pragma unroll
        for (int f = 0; f < 8; ++f)
#pragma unroll
            for (int r = 0; r < 8; ++r)
                acc[f][r] *= corr[r];

        // wave-internal cross-lane hand-off through LDS: drain DS stores
        asm volatile("s_wait_dscnt 0" ::: "memory");

        // reload P as two A-fragments (keys 0-31 and 32-63 of the step)
        const __bf16* pr = pw + l16 * LDP + 16 * h;
        v8bf p0lo = *(const v8bf*)pr;
        v8bf p0hi = *(const v8bf*)(pr + 8);
        v8bf p1lo = *(const v8bf*)(pr + 32);
        v8bf p1hi = *(const v8bf*)(pr + 40);
        v16bf pa0, pa1;
#pragma unroll
        for (int i = 0; i < 8; ++i) {
            pa0[i] = p0lo[i]; pa0[8 + i] = p0hi[i];
            pa1[i] = p1lo[i]; pa1[8 + i] = p1hi[i];
        }

        // row sums of (bf16-rounded) P via WMMA: P @ ones
        v8f rs = __builtin_amdgcn_wmma_f32_16x16x32_bf16(false, pa0, false, onesb,
                                                         (short)0, vzero,
                                                         false, false);
        rs = __builtin_amdgcn_wmma_f32_16x16x32_bf16(false, pa1, false, onesb,
                                                     (short)0, rs, false, false);
#pragma unroll
        for (int r = 0; r < 8; ++r)
            rsum[r] = rsum[r] * corr[r] + rs[r];

        // PV: acc[f](16q x 16c) += P(16q x 64k) * V(64k x 16c)
#pragma unroll
        for (int f = 0; f < 8; ++f) {
            const __bf16* vp = vbase + (size_t)(cbase + f * 16 + l16) * N_PIX
                             + kb + 16 * h;
            v16bf vf0 = *(const v16bf*)vp;
            v16bf vf1 = *(const v16bf*)(vp + 32);
            acc[f] = __builtin_amdgcn_wmma_f32_16x16x32_bf16(false, pa0, false, vf0,
                                                             (short)0, acc[f],
                                                             false, false);
            acc[f] = __builtin_amdgcn_wmma_f32_16x16x32_bf16(false, pa1, false, vf1,
                                                             (short)0, acc[f],
                                                             false, false);
        }
    }

    // epilogue: normalize, gamma*out + x
    const float g = gamma[0];
#pragma unroll
    for (int f = 0; f < 8; ++f) {
        int c = cbase + f * 16 + l16;
#pragma unroll
        for (int r = 0; r < 8; ++r) {
            int m = m0 + r + 8 * h;
            size_t idx = ((size_t)b * C_DIM + c) * N_PIX + m;
            out[idx] = g * (acc[f][r] / rsum[r]) + x[idx];
        }
    }
}

// ---------------------------------------------------------------------------
extern "C" void kernel_launch(void* const* d_in, const int* in_sizes, int n_in,
                              void* d_out, int out_size, void* d_ws, size_t ws_size,
                              hipStream_t stream) {
    const float* x     = (const float*)d_in[0];
    const float* Wq    = (const float*)d_in[1];
    const float* bq    = (const float*)d_in[2];
    const float* Wk    = (const float*)d_in[3];
    const float* bk    = (const float*)d_in[4];
    const float* Wv    = (const float*)d_in[5];
    const float* bv    = (const float*)d_in[6];
    const float* gamma = (const float*)d_in[7];
    float* out = (float*)d_out;

    // workspace layout (bf16):
    //   qT | kT : B*N*32       (1 MB each)
    //   vB      : B*C*N        (8 MB)
    //   xT      : B*N*C        (8 MB)
    //   Wqb|Wkb : 32*256       (16 KB each)
    //   Wvb     : 256*256      (128 KB)
    const size_t qk_elems = (size_t)B_DIM * N_PIX * CQK;
    const size_t cn_elems = (size_t)B_DIM * C_DIM * N_PIX;
    __bf16* qT  = (__bf16*)d_ws;
    __bf16* kT  = qT  + qk_elems;
    __bf16* vB  = kT  + qk_elems;
    __bf16* xT  = vB  + cn_elems;
    __bf16* Wqb = xT  + cn_elems;
    __bf16* Wkb = Wqb + (size_t)CQK * C_DIM;
    __bf16* Wvb = Wkb + (size_t)CQK * C_DIM;

    convert_w_kernel<<<dim3((C_DIM * C_DIM) / 256), 256, 0, stream>>>(
        Wq, Wk, Wv, Wqb, Wkb, Wvb);
    transpose_x_kernel<<<dim3(N_PIX / 32, C_DIM / 32, B_DIM), 256, 0, stream>>>(
        x, xT);
    proj_wmma_kernel<<<dim3(N_PIX / 128, B_DIM), 256, 0, stream>>>(
        xT, Wqb, bq, Wkb, bk, Wvb, bv, qT, kT, vB);
    attn_kernel<<<dim3(N_PIX / 16, B_DIM), 64, 0, stream>>>(
        qT, kT, vB, x, gamma, out);
}
